// GatedDeltaNetAttention_14456859918521
// MI455X (gfx1250) — matrix operand order repaired and checked by reading, not compiled, not run on hardware
//
#include <hip/hip_runtime.h>

// ---------------- problem constants ----------------
#define B_    2
#define T_    2048
#define C_    1024
#define H_    16
#define HKV_  4
#define D_    64
#define KC_   4
#define NREP_ (H_ / HKV_)
#define M_    (B_ * T_)      // 4096 rows

typedef unsigned short ushort_t;
typedef __attribute__((ext_vector_type(16))) __bf16 v16bf;
typedef __attribute__((ext_vector_type(8)))  float  v8f;

// ---------------- helpers ----------------
__device__ __forceinline__ ushort_t f2bf(float x) {
  unsigned u = __float_as_uint(x);
  u += 0x7FFFu + ((u >> 16) & 1u);   // round-to-nearest-even
  return (ushort_t)(u >> 16);
}

// A fragment (16x32 bf16 MxK) from LDS tile lA[128][32] row-major.
// lane(0-15)=M row; VGPR i: K = (i/4)*16 + (lane/16)*8 + (i%4)*2
// -> two contiguous 16B chunks per lane.
__device__ __forceinline__ v16bf lds_a_frag(const ushort_t* __restrict__ lA,
                                            int m0, int lane) {
  union { v16bf v; uint4 q[2]; } f;
  const ushort_t* row = lA + (m0 + (lane & 15)) * 32 + ((lane >> 4) << 3);
  f.q[0] = *(const uint4*)(row);
  f.q[1] = *(const uint4*)(row + 16);
  return f.v;
}

// B fragment (32x16 bf16 KxN) from LDS tile lB[32][128] row-major.
// lane(0-15)=N col; element s: K = (s/8)*16 + (lane/16)*8 + s%8 (strided gather).
__device__ __forceinline__ v16bf lds_b_frag(const ushort_t* __restrict__ lB,
                                            int n0, int lane) {
  union { v16bf v; ushort_t u[16]; } f;
  const ushort_t* col = lB + n0 + (lane & 15);
  const int kb = (lane >> 4) << 3;
#pragma unroll
  for (int s = 0; s < 16; ++s) f.u[s] = col[(((s >> 3) << 4) + kb + (s & 7)) * 128];
  return f.v;
}

// ---------------- conversion kernels ----------------
__global__ void gdn_cvt_bf16(const float* __restrict__ in, ushort_t* __restrict__ out,
                             unsigned n) {
  unsigned i = blockIdx.x * blockDim.x + threadIdx.x;
  if (i < n) out[i] = f2bf(in[i]);
}

// W is (N rows, K cols) row-major f32 -> Wt (K rows, N cols) bf16
__global__ void gdn_cvt_w_t(const float* __restrict__ W, ushort_t* __restrict__ Wt,
                            int N, int K) {
  unsigned i = blockIdx.x * blockDim.x + threadIdx.x;
  if (i < (unsigned)(N * K)) {
    int n = i / K, k = i % K;
    Wt[(size_t)k * N + n] = f2bf(W[i]);
  }
}

// ---------------- WMMA GEMM: C(M,N) = A(M,K) x Wt(K,N) ----------------
// block tile 128x128, 8 waves -> wave tile 32x64 (2x4 fragments of 16x16).
// K-slabs of 32 are double-buffered in LDS via async-to-LDS copies (ASYNCcnt).
// Dynamic LDS layout per buffer (16 KB): A[128][32]bf16 (8KB) + B[32][128]bf16 (8KB).
__global__ __launch_bounds__(256) void gdn_gemm_bf16(
    const ushort_t* __restrict__ A, const ushort_t* __restrict__ Bt,
    float* __restrict__ Cmat, int Mdim, int Ndim, int Kdim) {
  extern __shared__ char smem[];
  const int lane = threadIdx.x & 31;
  const int wave = threadIdx.x >> 5;
  const int bm = blockIdx.x * 128;
  const int bn = blockIdx.y * 128;
  const int wm = (wave & 3) * 32;        // local wave row
  const int wn = (wave >> 2) * 64;       // local wave col
  const unsigned lds0 = __builtin_amdgcn_groupstaticsize();  // dyn-LDS base offset

  // issue one 32-wide K-slab into LDS buffer `buf` (4 x b128 async loads / thread)
  auto issue_slab = [&](int buf, int k0) {
    const unsigned base = lds0 + (unsigned)buf * 16384u;
    const int tid = threadIdx.x;
#pragma unroll
    for (int r = 0; r < 2; ++r) {
      const int c = tid + (r << 8);                 // 0..511 chunk id
      {  // A chunk: row = c/4, 16B col-chunk = c%4
        const char* g = (const char*)A +
            (((size_t)(bm + (c >> 2)) * Kdim + k0) << 1) + ((c & 3) << 4);
        unsigned l = base + ((unsigned)c << 4);
        asm volatile("global_load_async_to_lds_b128 %0, %1, off"
                     :: "v"(l), "v"(g) : "memory");
      }
      {  // B chunk: row = c/16, 16B col-chunk = c%16
        const char* g = (const char*)Bt +
            (((size_t)(k0 + (c >> 4)) * Ndim + bn) << 1) + ((c & 15) << 4);
        unsigned l = base + 8192u + ((unsigned)c << 4);
        asm volatile("global_load_async_to_lds_b128 %0, %1, off"
                     :: "v"(l), "v"(g) : "memory");
      }
    }
  };

  v8f acc[2][4];
#pragma unroll
  for (int i = 0; i < 2; ++i)
#pragma unroll
    for (int j = 0; j < 4; ++j)
#pragma unroll
      for (int r = 0; r < 8; ++r) acc[i][j][r] = 0.0f;

  const int nslab = Kdim >> 5;
  issue_slab(0, 0);
  for (int s = 0; s < nslab; ++s) {
    const int cur = s & 1;
    asm volatile("s_wait_asynccnt 0x0" ::: "memory");  // my slab landed in LDS
    __syncthreads();                                   // everyone's slab landed
    if (s + 1 < nslab) issue_slab(cur ^ 1, (s + 1) << 5);

    const ushort_t* lA = (const ushort_t*)(smem + cur * 16384);
    const ushort_t* lB = (const ushort_t*)(smem + cur * 16384 + 8192);
    v16bf a0 = lds_a_frag(lA, wm,      lane);
    v16bf a1 = lds_a_frag(lA, wm + 16, lane);
    v16bf b0 = lds_b_frag(lB, wn,      lane);
    v16bf b1 = lds_b_frag(lB, wn + 16, lane);
    v16bf b2 = lds_b_frag(lB, wn + 32, lane);
    v16bf b3 = lds_b_frag(lB, wn + 48, lane);

    acc[0][0] = __builtin_amdgcn_wmma_f32_16x16x32_bf16(false, a0, false, b0, (short)0, acc[0][0], false, false);
    acc[0][1] = __builtin_amdgcn_wmma_f32_16x16x32_bf16(false, a0, false, b1, (short)0, acc[0][1], false, false);
    acc[0][2] = __builtin_amdgcn_wmma_f32_16x16x32_bf16(false, a0, false, b2, (short)0, acc[0][2], false, false);
    acc[0][3] = __builtin_amdgcn_wmma_f32_16x16x32_bf16(false, a0, false, b3, (short)0, acc[0][3], false, false);
    acc[1][0] = __builtin_amdgcn_wmma_f32_16x16x32_bf16(false, a1, false, b0, (short)0, acc[1][0], false, false);
    acc[1][1] = __builtin_amdgcn_wmma_f32_16x16x32_bf16(false, a1, false, b1, (short)0, acc[1][1], false, false);
    acc[1][2] = __builtin_amdgcn_wmma_f32_16x16x32_bf16(false, a1, false, b2, (short)0, acc[1][2], false, false);
    acc[1][3] = __builtin_amdgcn_wmma_f32_16x16x32_bf16(false, a1, false, b3, (short)0, acc[1][3], false, false);
  }

  // C/D layout: VGPR r -> M = r + 8*(lane>=16), N = lane%16
#pragma unroll
  for (int i = 0; i < 2; ++i)
#pragma unroll
    for (int j = 0; j < 4; ++j)
#pragma unroll
      for (int r = 0; r < 8; ++r) {
        int m = bm + wm + 16 * i + r + ((lane >> 4) << 3);
        int n = bn + wn + 16 * j + (lane & 15);
        Cmat[(size_t)m * Ndim + n] = acc[i][j][r];
      }
}

// ---------------- beta / gk micro-projection ----------------
__global__ __launch_bounds__(128) void gdn_beta_gk(
    const float* __restrict__ x, const float* __restrict__ Wbeta,
    const float* __restrict__ b_beta, const float* __restrict__ Wgk,
    const float* __restrict__ A_log, const float* __restrict__ dt_bias,
    float* __restrict__ beta, float* __restrict__ gk) {
  const int m = blockIdx.x;
  const int hkv = threadIdx.x >> 5;
  const int lane = threadIdx.x & 31;
  const float* xr = x + (size_t)m * C_;
  float db = 0.f, dg = 0.f;
  for (int i = lane; i < C_; i += 32) {
    float xv = xr[i];
    db = fmaf(xv, Wbeta[hkv * C_ + i], db);
    dg = fmaf(xv, Wgk[hkv * C_ + i], dg);
  }
#pragma unroll
  for (int o = 16; o; o >>= 1) { db += __shfl_xor(db, o); dg += __shfl_xor(dg, o); }
  if (lane == 0) {
    float bb = db + b_beta[hkv];
    beta[m * HKV_ + hkv] = 1.f / (1.f + __expf(-bb));
    float gz = dg + dt_bias[hkv];
    float sp = (gz > 20.f) ? gz : log1pf(__expf(gz));
    gk[m * HKV_ + hkv] = -__expf(A_log[hkv]) * sp;
  }
}

// ---------------- depthwise causal conv (+ optional per-head l2norm) ----------------
__global__ __launch_bounds__(64) void gdn_conv_norm(
    const float* __restrict__ in, const float* __restrict__ cw,
    float* __restrict__ out, int nh, int do_norm) {
  const int bt = blockIdx.x / nh;
  const int h  = blockIdx.x % nh;
  const int t  = bt % T_;
  const int e  = threadIdx.x;
  const int c  = h * D_ + e;
  const int nc = nh * D_;
  float acc = 0.f;
#pragma unroll
  for (int j = 0; j < KC_; ++j) {
    int tt = t - (KC_ - 1) + j;
    if (tt >= 0)
      acc = fmaf(cw[c * KC_ + j], in[(size_t)(bt - (KC_ - 1) + j) * nc + c], acc);
  }
  float val = acc;
  if (do_norm) {
    float ss = val * val;
#pragma unroll
    for (int o = 16; o; o >>= 1) ss += __shfl_xor(ss, o);
    __shared__ float red[2];
    if ((e & 31) == 0) red[e >> 5] = ss;
    __syncthreads();
    val *= rsqrtf(red[0] + red[1] + 1e-12f);
  }
  out[(size_t)bt * nc + c] = val;
}

// ---------------- gated delta-rule recurrence ----------------
// one block per (b,h); lane e owns state column S[:,e] in 64 VGPRs.
// k/q broadcasts double-buffered in LDS -> one barrier per time step.
__global__ __launch_bounds__(64) void gdn_recurrence(
    const float* __restrict__ qn, const float* __restrict__ kn,
    const float* __restrict__ vc, const float* __restrict__ beta,
    const float* __restrict__ gk, float* __restrict__ o) {
  const int bh = blockIdx.x;
  const int b = bh / H_, h = bh % H_, hkv = h / NREP_;
  const int e = threadIdx.x;
  float S[D_];
#pragma unroll
  for (int d = 0; d < D_; ++d) S[d] = 0.f;
  __shared__ float sk[2][D_], sq[2][D_];
  {
    const size_t bt0 = (size_t)b * T_;
    sk[0][e] = kn[(bt0 * HKV_ + hkv) * D_ + e];
    sq[0][e] = qn[(bt0 * H_  + h  ) * D_ + e];
  }
  for (int t = 0; t < T_; ++t) {
    const int cur = t & 1;
    const size_t bt = (size_t)b * T_ + t;
    __syncthreads();                       // cur buffer complete, prev reads done
    if (t + 1 < T_) {
      sk[cur ^ 1][e] = kn[((bt + 1) * HKV_ + hkv) * D_ + e];
      sq[cur ^ 1][e] = qn[((bt + 1) * H_  + h  ) * D_ + e];
    }
    const float ve  = vc[(bt * HKV_ + hkv) * D_ + e];
    const float btv = beta[bt * HKV_ + hkv];
    const float gt  = __expf(gk[bt * HKV_ + hkv]);
    float ks = 0.f;
#pragma unroll
    for (int d = 0; d < D_; ++d) { S[d] *= gt; ks = fmaf(sk[cur][d], S[d], ks); }
    const float upd = btv * (ve - ks);
    float oe = 0.f;
#pragma unroll
    for (int d = 0; d < D_; ++d) { S[d] = fmaf(sk[cur][d], upd, S[d]); oe = fmaf(sq[cur][d], S[d], oe); }
    o[(bt * H_ + h) * D_ + e] = oe * 0.125f;   // D^-0.5
  }
}

// ---------------- per-head RMSNorm * SiLU gate -> bf16 ----------------
__global__ __launch_bounds__(64) void gdn_gate_rms(
    const float* __restrict__ o, const float* __restrict__ gate,
    const float* __restrict__ rms_w, ushort_t* __restrict__ og) {
  const int bth = blockIdx.x;            // (b*T+t)*H + h
  const int e = threadIdx.x;
  const size_t idx = (size_t)bth * D_ + e;
  float v = o[idx];
  float ss = v * v;
#pragma unroll
  for (int off = 16; off; off >>= 1) ss += __shfl_xor(ss, off);
  __shared__ float red[2];
  if ((e & 31) == 0) red[e >> 5] = ss;
  __syncthreads();
  float ms = (red[0] + red[1]) * (1.0f / D_);
  float nv = v * rsqrtf(ms + 1e-5f) * rms_w[e];
  float gx = gate[idx];
  float g = gx / (1.f + __expf(-gx));    // silu
  og[idx] = f2bf(nv * g);
}

// ---------------- launch ----------------
extern "C" void kernel_launch(void* const* d_in, const int* in_sizes, int n_in,
                              void* d_out, int out_size, void* d_ws, size_t ws_size,
                              hipStream_t stream) {
  const float* x      = (const float*)d_in[0];
  const float* Wq     = (const float*)d_in[1];
  const float* Wk     = (const float*)d_in[2];
  const float* Wv     = (const float*)d_in[3];
  const float* Wproj  = (const float*)d_in[4];
  const float* cwq    = (const float*)d_in[5];
  const float* cwk    = (const float*)d_in[6];
  const float* cwv    = (const float*)d_in[7];
  const float* Wbeta  = (const float*)d_in[8];
  const float* b_beta = (const float*)d_in[9];
  const float* Wgk    = (const float*)d_in[10];
  const float* A_log  = (const float*)d_in[11];
  const float* dt_b   = (const float*)d_in[12];
  const float* Wgate  = (const float*)d_in[13];
  const float* rms_w  = (const float*)d_in[14];
  float* out = (float*)d_out;

  char* w = (char*)d_ws;
  size_t off = 0;
  auto carve = [&](size_t bytes) -> void* {
    void* p = w + off;
    off += (bytes + 255) & ~(size_t)255;
    return p;
  };
  ushort_t* x_bf     = (ushort_t*)carve((size_t)M_ * C_ * 2);
  ushort_t* Wq_t     = (ushort_t*)carve((size_t)C_ * (H_ * D_) * 2);
  ushort_t* Wk_t     = (ushort_t*)carve((size_t)C_ * (HKV_ * D_) * 2);
  ushort_t* Wv_t     = (ushort_t*)carve((size_t)C_ * (HKV_ * D_) * 2);
  ushort_t* Wgate_t  = (ushort_t*)carve((size_t)C_ * (H_ * D_) * 2);
  ushort_t* Wproj_t  = (ushort_t*)carve((size_t)C_ * C_ * 2);
  float* q_proj      = (float*)carve((size_t)M_ * (H_ * D_) * 4);
  float* k_proj      = (float*)carve((size_t)M_ * (HKV_ * D_) * 4);
  float* v_proj      = (float*)carve((size_t)M_ * (HKV_ * D_) * 4);
  float* gate_proj   = (float*)carve((size_t)M_ * (H_ * D_) * 4);
  float* q_n         = (float*)carve((size_t)M_ * (H_ * D_) * 4);
  float* k_n         = (float*)carve((size_t)M_ * (HKV_ * D_) * 4);
  float* v_c         = (float*)carve((size_t)M_ * (HKV_ * D_) * 4);
  float* beta        = (float*)carve((size_t)M_ * HKV_ * 4);
  float* gk          = (float*)carve((size_t)M_ * HKV_ * 4);
  float* o_buf       = (float*)carve((size_t)M_ * (H_ * D_) * 4);
  ushort_t* og_bf    = (ushort_t*)carve((size_t)M_ * (H_ * D_) * 2);

  // 1) precision conversion + weight transposes
  gdn_cvt_bf16<<<(M_ * C_ + 255) / 256, 256, 0, stream>>>(x, x_bf, M_ * C_);
  gdn_cvt_w_t<<<((H_ * D_) * C_ + 255) / 256, 256, 0, stream>>>(Wq, Wq_t, H_ * D_, C_);
  gdn_cvt_w_t<<<((HKV_ * D_) * C_ + 255) / 256, 256, 0, stream>>>(Wk, Wk_t, HKV_ * D_, C_);
  gdn_cvt_w_t<<<((HKV_ * D_) * C_ + 255) / 256, 256, 0, stream>>>(Wv, Wv_t, HKV_ * D_, C_);
  gdn_cvt_w_t<<<((H_ * D_) * C_ + 255) / 256, 256, 0, stream>>>(Wgate, Wgate_t, H_ * D_, C_);
  gdn_cvt_w_t<<<(C_ * C_ + 255) / 256, 256, 0, stream>>>(Wproj, Wproj_t, C_, C_);

  // 2) WMMA projections (32 KB dynamic LDS: 2 x (8KB A + 8KB B))
  const size_t lds = 32768;
  gdn_gemm_bf16<<<dim3(M_ / 128, (H_ * D_) / 128), 256, lds, stream>>>(x_bf, Wq_t, q_proj, M_, H_ * D_, C_);
  gdn_gemm_bf16<<<dim3(M_ / 128, (HKV_ * D_) / 128), 256, lds, stream>>>(x_bf, Wk_t, k_proj, M_, HKV_ * D_, C_);
  gdn_gemm_bf16<<<dim3(M_ / 128, (HKV_ * D_) / 128), 256, lds, stream>>>(x_bf, Wv_t, v_proj, M_, HKV_ * D_, C_);
  gdn_gemm_bf16<<<dim3(M_ / 128, (H_ * D_) / 128), 256, lds, stream>>>(x_bf, Wgate_t, gate_proj, M_, H_ * D_, C_);

  // 3) beta / decay micro-projection (f32 for accuracy)
  gdn_beta_gk<<<M_, 128, 0, stream>>>(x, Wbeta, b_beta, Wgk, A_log, dt_b, beta, gk);

  // 4) causal depthwise conv + l2norm
  gdn_conv_norm<<<M_ * H_,   64, 0, stream>>>(q_proj, cwq, q_n, H_,   1);
  gdn_conv_norm<<<M_ * HKV_, 64, 0, stream>>>(k_proj, cwk, k_n, HKV_, 1);
  gdn_conv_norm<<<M_ * HKV_, 64, 0, stream>>>(v_proj, cwv, v_c, HKV_, 0);

  // 5) sequential gated delta-rule (32 independent chains)
  gdn_recurrence<<<B_ * H_, 64, 0, stream>>>(q_n, k_n, v_c, beta, gk, o_buf);

  // 6) RMSNorm * SiLU gate
  gdn_gate_rms<<<M_ * H_, 64, 0, stream>>>(o_buf, gate_proj, rms_w, og_bf);

  // 7) output projection (WMMA) -> d_out (f32)
  gdn_gemm_bf16<<<dim3(M_ / 128, C_ / 128), 256, lds, stream>>>(og_bf, Wproj_t, out, M_, C_, C_);
}
